// Mamba2Block_66803921322400
// MI455X (gfx1250) — matrix operand order repaired and checked by reading, not compiled, not run on hardware
//
#include <hip/hip_runtime.h>
#include <hip/hip_bf16.h>

typedef __attribute__((ext_vector_type(16))) _Float16 v16h;
typedef __attribute__((ext_vector_type(2)))  _Float16 h2;
typedef __attribute__((ext_vector_type(8)))  float    v8f;

// Problem constants (from reference)
constexpr int LSEQ  = 1024;
constexpr int DM    = 512;
constexpr int HH    = 16;
constexpr int PP    = 64;
constexpr int NN    = 64;
constexpr int CSZ   = 256;
constexpr int NC    = LSEQ / CSZ;               // 4
constexpr int DPROJ = 2 * HH * PP + 2 * NN + 1; // 2177
constexpr int HP    = HH * PP;                  // 1024

// GEMM tiling
#define TM 64
#define TN 64
#define KB 32
#define LPAD 8

struct GemmP {
  const float* A; const float* B; float* C;
  const float* bias; const float* rowscale;
  int M, N, K, lda, ldb, ldc;
  long sA1, sA2, sB1, sB2, sC1, sC2, sR1;
  int nz2;
};

// Generic f32-in / f16-WMMA / f32-out GEMM, guard-free (all dims exact tile
// multiples).  C = [C +] rowscale(row) * (A@B) [+ bias]
template <int TA, int TB, int BIAS, int RS, int ACC>
__global__ __launch_bounds__(256)
void k_gemm(GemmP p) {
  int z  = blockIdx.z;
  int z1 = z / p.nz2, z2 = z % p.nz2;
  const float* __restrict__ A = p.A + (long)z1 * p.sA1 + (long)z2 * p.sA2;
  const float* __restrict__ B = p.B + (long)z1 * p.sB1 + (long)z2 * p.sB2;
  float*       __restrict__ C = p.C + (long)z1 * p.sC1 + (long)z2 * p.sC2;
  const float* __restrict__ rs = RS ? (p.rowscale + (long)z1 * p.sR1) : nullptr;

  __shared__ _Float16 As[TM][KB + LPAD];   // [m][k]
  __shared__ _Float16 Bs[TN][KB + LPAD];   // [n][k] (transposed stage)

  int tid  = threadIdx.x;
  int lane = tid & 31;
  int wave = tid >> 5;
  int bm = blockIdx.y * TM;
  int bn = blockIdx.x * TN;
  int mt = (wave & 3) * 16;
  int nt = (wave >> 2) * 32;

  v8f acc0 = {};
  v8f acc1 = {};

  for (int k0 = 0; k0 < p.K; k0 += KB) {
    // ---- stage A tile (f32 -> f16); each thread packs a k-pair -> b32 DS store
    if (!TA) {
#pragma unroll
      for (int i = 0; i < (TM * KB / 2) / 256; ++i) {
        int idx = tid + i * 256;
        int m = idx >> 4, k = (idx & 15) * 2;        // 16 k-pairs per row
        const float* src = &A[(bm + m) * p.lda + (k0 + k)];
        h2 hv = { (_Float16)src[0], (_Float16)src[1] };
        *(h2*)&As[m][k] = hv;
      }
    } else {
#pragma unroll
      for (int i = 0; i < (TM * KB / 2) / 256; ++i) {
        int idx = tid + i * 256;
        int m = idx & 63, k = (idx >> 6) * 2;
        h2 hv = { (_Float16)A[(k0 + k) * p.lda + (bm + m)],
                  (_Float16)A[(k0 + k + 1) * p.lda + (bm + m)] };
        *(h2*)&As[m][k] = hv;
      }
    }
    // ---- stage B tile transposed into [n][k] ----
    if (!TB) {
#pragma unroll
      for (int i = 0; i < (TN * KB / 2) / 256; ++i) {
        int idx = tid + i * 256;
        int n = idx & 63, k = (idx >> 6) * 2;
        h2 hv = { (_Float16)B[(k0 + k) * p.ldb + (bn + n)],
                  (_Float16)B[(k0 + k + 1) * p.ldb + (bn + n)] };
        *(h2*)&Bs[n][k] = hv;
      }
    } else {
#pragma unroll
      for (int i = 0; i < (TN * KB / 2) / 256; ++i) {
        int idx = tid + i * 256;
        int n = idx >> 4, k = (idx & 15) * 2;
        const float* src = &B[(bn + n) * p.ldb + (k0 + k)];
        h2 hv = { (_Float16)src[0], (_Float16)src[1] };
        *(h2*)&Bs[n][k] = hv;
      }
    }
    __syncthreads();

    // ---- fragments: A and B both use contiguous k = 16*g + s ----
    int g  = lane >> 4;
    int mr = mt + (lane & 15);
    v16h a;
#pragma unroll
    for (int v = 0; v < 4; ++v) {
      a[2 * v]         = As[mr][16 * g + 2 * v];
      a[2 * v + 1]     = As[mr][16 * g + 2 * v + 1];
      a[8 + 2 * v]     = As[mr][16 * g + 8 + 2 * v];
      a[8 + 2 * v + 1] = As[mr][16 * g + 8 + 2 * v + 1];
    }
    int nr0 = nt + (lane & 15);
    int nr1 = nr0 + 16;
    v16h b0, b1;
#pragma unroll
    for (int v = 0; v < 8; ++v) {
      b0[2 * v]     = Bs[nr0][16 * g + 2 * v];
      b0[2 * v + 1] = Bs[nr0][16 * g + 2 * v + 1];
      b1[2 * v]     = Bs[nr1][16 * g + 2 * v];
      b1[2 * v + 1] = Bs[nr1][16 * g + 2 * v + 1];
    }
    acc0 = __builtin_amdgcn_wmma_f32_16x16x32_f16(false, a, false, b0, (short)0, acc0, false, false);
    acc1 = __builtin_amdgcn_wmma_f32_16x16x32_f16(false, a, false, b1, (short)0, acc1, false, false);
    __syncthreads();
  }

  // ---- epilogue (C/D layout: row = v + 8*g, col = lane%16) ----
  int g    = lane >> 4;
  int col0 = bn + nt + (lane & 15);
  int col1 = col0 + 16;
  int rbase = bm + mt + 8 * g;
  float bias0 = BIAS ? p.bias[col0] : 0.f;
  float bias1 = BIAS ? p.bias[col1] : 0.f;

  float old0[8], old1[8];
  if (ACC) {
#pragma unroll
    for (int v = 0; v < 8; ++v) {
      old0[v] = C[(rbase + v) * p.ldc + col0];
      old1[v] = C[(rbase + v) * p.ldc + col1];
    }
  }
#pragma unroll
  for (int v = 0; v < 8; ++v) {
    int row = rbase + v;
    float sc = RS ? rs[row] : 1.f;
    float v0 = acc0[v] * sc + bias0;
    float v1 = acc1[v] * sc + bias1;
    if (ACC) { v0 += old0[v]; v1 += old1[v]; }
    C[row * p.ldc + col0] = v0;
    C[row * p.ldc + col1] = v1;
  }
}

// Last proj column (dt_raw): proj[t][2176] = u[t]·Wi[:,2176] + bi[2176]
__global__ __launch_bounds__(256)
void k_lastcol(const float* __restrict__ u, const float* __restrict__ Wi,
               const float* __restrict__ bi, float* __restrict__ proj) {
  int wave = threadIdx.x >> 5;
  int lane = threadIdx.x & 31;
  int row  = blockIdx.x * 8 + wave;
  float s = 0.f;
#pragma unroll 4
  for (int k = lane; k < DM; k += 32)
    s += u[row * DM + k] * Wi[k * DPROJ + (DPROJ - 1)];
#pragma unroll
  for (int off = 16; off > 0; off >>= 1)
    s += __shfl_down(s, off, 32);
  if (lane == 0) proj[row * DPROJ + (DPROJ - 1)] = s + bi[DPROJ - 1];
}

// dt = softplus(proj[:, 2176]); Bn/Cn = rmsnorm over N=64 (* wB / wC)
__global__ __launch_bounds__(64)
void k_postproj(const float* __restrict__ proj, const float* __restrict__ wB,
                const float* __restrict__ wC, float* __restrict__ dt,
                float* __restrict__ Bn, float* __restrict__ Cn) {
  int t = blockIdx.x, j = threadIdx.x;
  const float* row = proj + t * DPROJ;
  if (j == 0) {
    float x = row[2 * HP + 2 * NN];
    dt[t] = (x > 20.f) ? x : log1pf(expf(x));
  }
  float b = row[2 * HP + j];
  float c = row[2 * HP + NN + j];
  __shared__ float sb[64], sc2[64];
  sb[j] = b * b; sc2[j] = c * c;
  __syncthreads();
  for (int off = 32; off > 0; off >>= 1) {
    if (j < off) { sb[j] += sb[j + off]; sc2[j] += sc2[j + off]; }
    __syncthreads();
  }
  float rb = rsqrtf(sb[0] * (1.f / 64.f) + 1e-5f);
  float rc = rsqrtf(sc2[0] * (1.f / 64.f) + 1e-5f);
  Bn[t * NN + j] = b * rb * wB[j];
  Cn[t * NN + j] = c * rc * wC[j];
}

// Per-chunk inclusive scan of la = dt*(-exp(A_log)); derived decay quantities.
__global__ __launch_bounds__(256)
void k_scan(const float* __restrict__ dt, const float* __restrict__ A_log,
            float* __restrict__ cs, float* __restrict__ dintra,
            float* __restrict__ wdec, float* __restrict__ cdec) {
  int c = blockIdx.x, s = threadIdx.x;
  int t = c * CSZ + s;
  float Aneg = -expf(A_log[0]);
  float la = dt[t] * Aneg;
  __shared__ float sh[CSZ];
  sh[s] = la;
  __syncthreads();
  for (int off = 1; off < CSZ; off <<= 1) {
    float v = sh[s];
    if (s >= off) v += sh[s - off];
    __syncthreads();
    sh[s] = v;
    __syncthreads();
  }
  float csv = sh[s];
  float cl  = sh[CSZ - 1];
  cs[t]     = csv;
  dintra[t] = expf(csv);
  wdec[t]   = expf(cl - csv) * dt[t];
  if (s == 0) cdec[c] = expf(cl);
}

// Bw[t][n] = wdec[t] * Bn[t][n]
__global__ __launch_bounds__(256)
void k_bw(const float* __restrict__ wdec, const float* __restrict__ Bn,
          float* __restrict__ Bw) {
  int idx = blockIdx.x * 256 + threadIdx.x;
  int t = idx >> 6;
  Bw[idx] = wdec[t] * Bn[idx];
}

// M[c][t][s] = tril ? exp(cs[t]-cs[s]) * dt[s] * (C·B^T)[t][s] : 0   (in place)
__global__ __launch_bounds__(256)
void k_maskM(const float* __restrict__ cs, const float* __restrict__ dt,
             float* __restrict__ Mbuf) {
  int c = blockIdx.y;
  int idx = blockIdx.x * 256 + threadIdx.x;   // 65536 per chunk
  int tt = idx >> 8, ss = idx & 255;
  int o = c * CSZ * CSZ + idx;
  float gcb = Mbuf[o];
  int gt = c * CSZ + tt, gs = c * CSZ + ss;
  Mbuf[o] = (ss <= tt) ? expf(cs[gt] - cs[gs]) * dt[gs] * gcb : 0.f;
}

// Inter-chunk recurrence: h_inter[c] = h_inter[c-1]*cdec[c-1] + h_chunk[c-1]
__global__ __launch_bounds__(256)
void k_rec(const float* __restrict__ h_chunk, const float* __restrict__ cdec,
           float* __restrict__ h_inter) {
  int j = blockIdx.x * 256 + threadIdx.x;    // 64*1024
  float h = 0.f;
#pragma unroll
  for (int c = 0; c < NC; ++c) {
    h_inter[c * (NN * HP) + j] = h;
    h = h * cdec[c] + h_chunk[c * (NN * HP) + j];
  }
}

// y = (y + x*D) * silu(z)   (in place on y_pre)
__global__ __launch_bounds__(256)
void k_gate(const float* __restrict__ proj, const float* __restrict__ D,
            float* __restrict__ y_pre) {
  int idx = blockIdx.x * 256 + threadIdx.x;  // 1024*1024
  int t = idx >> 10, j = idx & 1023;
  const float* row = proj + t * DPROJ;
  float zv = row[j];
  float xv = row[HP + j];
  float y  = y_pre[idx] + xv * D[j >> 6];
  float sig = 1.f / (1.f + expf(-zv));
  y_pre[idx] = y * zv * sig;
}

extern "C" void kernel_launch(void* const* d_in, const int* in_sizes, int n_in,
                              void* d_out, int out_size, void* d_ws, size_t ws_size,
                              hipStream_t stream) {
  (void)in_sizes; (void)n_in; (void)out_size; (void)ws_size;
  const float* u     = (const float*)d_in[0];
  const float* Wi    = (const float*)d_in[1];
  const float* bi    = (const float*)d_in[2];
  const float* A_log = (const float*)d_in[3];
  const float* Dv    = (const float*)d_in[4];
  const float* wB    = (const float*)d_in[5];
  const float* wC    = (const float*)d_in[6];
  const float* Wo    = (const float*)d_in[7];
  float* out = (float*)d_out;
  float* ws  = (float*)d_ws;

  // workspace layout (floats)
  float* proj    = ws;                                  // 1024*2177
  float* Bn      = proj    + (long)LSEQ * DPROJ;        // 65536
  float* Cn      = Bn      + (long)LSEQ * NN;           // 65536
  float* Bw      = Cn      + (long)LSEQ * NN;           // 65536
  float* dt      = Bw      + (long)LSEQ * NN;           // 1024
  float* cs      = dt      + LSEQ;                      // 1024
  float* dintra  = cs      + LSEQ;                      // 1024
  float* wdec    = dintra  + LSEQ;                      // 1024
  float* cdec    = wdec    + LSEQ;                      // 64 (4 used)
  float* Mbuf    = cdec    + 64;                        // 4*256*256
  float* h_chunk = Mbuf    + (long)NC * CSZ * CSZ;      // 4*64*1024
  float* h_inter = h_chunk + (long)NC * NN * HP;        // 4*64*1024
  float* y_pre   = h_inter + (long)NC * NN * HP;        // 1024*1024

  dim3 blk(256);

  // 1) proj[:, 0:2176] = u @ Wi[:, 0:2176] + bi   (guard-free, N=2176)
  {
    GemmP p{};
    p.A = u; p.B = Wi; p.C = proj; p.bias = bi;
    p.M = LSEQ; p.N = DPROJ - 1; p.K = DM;
    p.lda = DM; p.ldb = DPROJ; p.ldc = DPROJ;
    p.nz2 = 1;
    hipLaunchKernelGGL((k_gemm<0, 0, 1, 0, 0>),
                       dim3((DPROJ - 1) / TN, LSEQ / TM, 1), blk, 0, stream, p);
  }
  // 1b) last column (dt_raw)
  hipLaunchKernelGGL(k_lastcol, dim3(LSEQ / 8), blk, 0, stream, u, Wi, bi, proj);
  // 2) dt / rmsnorm
  hipLaunchKernelGGL(k_postproj, dim3(LSEQ), dim3(64), 0, stream, proj, wB, wC, dt, Bn, Cn);
  // 3) per-chunk scan
  hipLaunchKernelGGL(k_scan, dim3(NC), dim3(CSZ), 0, stream, dt, A_log, cs, dintra, wdec, cdec);
  // 4) Bw
  hipLaunchKernelGGL(k_bw, dim3((LSEQ * NN) / 256), blk, 0, stream, wdec, Bn, Bw);
  // 5) Gcb = Cn @ Bn^T per chunk
  {
    GemmP p{};
    p.A = Cn; p.B = Bn; p.C = Mbuf;
    p.M = CSZ; p.N = CSZ; p.K = NN;
    p.lda = NN; p.ldb = NN; p.ldc = CSZ;
    p.nz2 = 1;
    p.sA1 = (long)CSZ * NN; p.sB1 = (long)CSZ * NN; p.sC1 = (long)CSZ * CSZ;
    hipLaunchKernelGGL((k_gemm<0, 1, 0, 0, 0>),
                       dim3(CSZ / TN, CSZ / TM, NC), blk, 0, stream, p);
  }
  // 6) mask + decay
  hipLaunchKernelGGL(k_maskM, dim3((CSZ * CSZ) / 256, NC), blk, 0, stream, cs, dt, Mbuf);
  // 7) chunk states: h_chunk[c] = Bw[c]^T @ X[c]
  {
    GemmP p{};
    p.A = Bw; p.B = proj + HP; p.C = h_chunk;
    p.M = NN; p.N = HP; p.K = CSZ;
    p.lda = NN; p.ldb = DPROJ; p.ldc = HP;
    p.nz2 = 1;
    p.sA1 = (long)CSZ * NN; p.sB1 = (long)CSZ * DPROJ; p.sC1 = (long)NN * HP;
    hipLaunchKernelGGL((k_gemm<1, 0, 0, 0, 0>),
                       dim3(HP / TN, 1, NC), blk, 0, stream, p);
  }
  // 8) recurrence
  hipLaunchKernelGGL(k_rec, dim3((NN * HP) / 256), blk, 0, stream, h_chunk, cdec, h_inter);
  // 9) y_diag[c] = M[c] @ X[c]
  {
    GemmP p{};
    p.A = Mbuf; p.B = proj + HP; p.C = y_pre;
    p.M = CSZ; p.N = HP; p.K = CSZ;
    p.lda = CSZ; p.ldb = DPROJ; p.ldc = HP;
    p.nz2 = 1;
    p.sA1 = (long)CSZ * CSZ; p.sB1 = (long)CSZ * DPROJ; p.sC1 = (long)CSZ * HP;
    hipLaunchKernelGGL((k_gemm<0, 0, 0, 0, 0>),
                       dim3(HP / TN, CSZ / TM, NC), blk, 0, stream, p);
  }
  // 10) y_off[c,h]: y_pre += dintra[t] * (Cn[c] @ h_inter[c][:, h*64:])
  {
    GemmP p{};
    p.A = Cn; p.B = h_inter; p.C = y_pre;
    p.rowscale = dintra; p.sR1 = CSZ;
    p.M = CSZ; p.N = PP; p.K = NN;
    p.lda = NN; p.ldb = HP; p.ldc = HP;
    p.nz2 = HH;
    p.sA1 = (long)CSZ * NN; p.sA2 = 0;
    p.sB1 = (long)NN * HP;  p.sB2 = PP;
    p.sC1 = (long)CSZ * HP; p.sC2 = PP;
    hipLaunchKernelGGL((k_gemm<0, 0, 0, 1, 1>),
                       dim3(1, CSZ / TM, NC * HH), blk, 0, stream, p);
  }
  // 11) gate: (y + x*D) * silu(z)
  hipLaunchKernelGGL(k_gate, dim3((LSEQ * HP) / 256), blk, 0, stream, proj, Dv, y_pre);
  // 12) out = y_act @ Wo
  {
    GemmP p{};
    p.A = y_pre; p.B = Wo; p.C = out;
    p.M = LSEQ; p.N = DM; p.K = HP;
    p.lda = HP; p.ldb = DM; p.ldc = DM;
    p.nz2 = 1;
    hipLaunchKernelGGL((k_gemm<0, 0, 0, 0, 0>),
                       dim3(DM / TN, LSEQ / TM, 1), blk, 0, stream, p);
  }
}
